// BatchGraphEncoder_21646635172625
// MI455X (gfx1250) — matrix operand order repaired and checked by reading, not compiled, not run on hardware
//
#include <hip/hip_runtime.h>
#include <hip/hip_bf16.h>
#include <math.h>

// ---------------------------------------------------------------------------
// BatchGraphEncoder for MI455X (gfx1250):
//   wave32 + V_WMMA_F32_16X16X32_F16 + GLOBAL_LOAD_ASYNC_TO_LDS_B128
// ---------------------------------------------------------------------------

typedef _Float16 h16;
typedef __attribute__((ext_vector_type(16))) _Float16 v16h;
typedef __attribute__((ext_vector_type(8)))  _Float16 v8h;
typedef __attribute__((ext_vector_type(8)))  float    v8f;

#define NB 8
#define NA 64
#define NT 12
#define NH 128
#define NE 128
#define NTY 8
#define NSC 32
#define NAG 16
#define KEDGE 384   // [h_i | h_j | es]
#define KG    768   // [coord|type|node_edges|scene|agent|h]
#define SEDGE 392   // LDS row stride (halfs), mult of 8, odd*8 -> conflict-free
#define SX    776   // xrow stride (halfs)
#define SH    136   // 128-wide f16 tile stride (halfs)

__device__ __forceinline__ v16h cat8(v8h lo, v8h hi) {
  return __builtin_shufflevector(lo, hi, 0,1,2,3,4,5,6,7,8,9,10,11,12,13,14,15);
}

// A-fragment (16x32 f16) from row-major f16 LDS tile.
// lane<16: M=lane, K={0..7,16..23}; lane>=16: M=lane-16, K={8..15,24..31}
__device__ __forceinline__ v16h load_a_lds(const h16* lds, int stride, int row0, int k0) {
  int lane = threadIdx.x & 31;
  int m  = lane & 15;
  int hh = (lane >> 4) << 3;
  const h16* p = lds + (row0 + m) * stride + k0 + hh;
  v8h lo = *(const v8h*)(p);
  v8h hi = *(const v8h*)(p + 16);
  return cat8(lo, hi);
}

// B-fragment (32x16 f16) from N-major (transposed, NxK) f16 weights in global.
// lane<16: N=lane, K={0..7,16..23}; lane>=16: N=lane-16, K={8..15,24..31}
__device__ __forceinline__ v16h load_b_glb(const h16* Wt, int K, int n0, int k0) {
  int lane = threadIdx.x & 31;
  int n    = lane & 15;
  int base = k0 + ((lane >> 4) << 3);
  const h16* p = Wt + (size_t)(n0 + n) * K + base;
  v8h lo = *(const v8h*)(p);
  v8h hi = *(const v8h*)(p + 16);
  return cat8(lo, hi);
}

#define WMMA(a, b, c) \
  __builtin_amdgcn_wmma_f32_16x16x32_f16(false, (a), false, (b), (short)0, (c), false, false)

__device__ __forceinline__ float sigmf(float x) { return 1.0f / (1.0f + __expf(-x)); }

// ---------------------------------------------------------------------------
// Precompute / setup kernels (skinny-K: scalar VALU is fine)
// ---------------------------------------------------------------------------

__global__ void zero_state_kernel(float* hG, float* cG, float* seG) {
  int idx = blockIdx.x * 256 + threadIdx.x;
  if (idx >= NB * NA * NH) return;
  hG[idx] = 0.f; cG[idx] = 0.f; seG[idx] = 0.f;
}

// dst[n*K+k] = (h16) src[k*N+n]
__global__ void transpose_to_h16(const float* __restrict__ src, h16* __restrict__ dst,
                                 int K, int N) {
  int idx = blockIdx.x * 256 + threadIdx.x;
  if (idx >= K * N) return;
  int n = idx / K, k = idx - n * K;
  dst[idx] = (h16)src[(size_t)k * N + n];
}

// Combined gate weight, transposed: rows n=0..511, k<640 -> W_ih, else W_hh
__global__ void build_wgT(const float* __restrict__ W_ih, const float* __restrict__ W_hh,
                          h16* __restrict__ dst) {
  int idx = blockIdx.x * 256 + threadIdx.x;
  if (idx >= 512 * KG) return;
  int n = idx / KG, k = idx - n * KG;
  float v = (k < 640) ? W_ih[(size_t)k * 512 + n] : W_hh[(size_t)(k - 640) * 512 + n];
  dst[idx] = (h16)v;
}

__global__ void enc_coord_kernel(const float* __restrict__ ntraj, const float* __restrict__ traj,
                                 const float* __restrict__ W_in, const float* __restrict__ b_in,
                                 float* __restrict__ coordG) {
  int idx = blockIdx.x * 256 + threadIdx.x;              // (b,a,t,c)
  if (idx >= NB * NA * NT * NH) return;
  int c = idx & 127, row = idx >> 7;                     // row = (b*64+a)*12+t
  float s = b_in[c];
  #pragma unroll
  for (int k = 0; k < 3; ++k) s += ntraj[row * 3 + k] * W_in[k * NH + c];
  #pragma unroll
  for (int k = 0; k < 3; ++k) s += traj[row * 3 + k] * W_in[(3 + k) * NH + c];
  coordG[idx] = fmaxf(s, 0.f);
}

__global__ void enc_type_kernel(const float* __restrict__ atype,
                                const float* __restrict__ Wnt, const float* __restrict__ bnt,
                                float* __restrict__ typeG) {
  int idx = blockIdx.x * 256 + threadIdx.x;              // (b,a,c)
  if (idx >= NB * NA * NH) return;
  int c = idx & 127, row = idx >> 7;
  float s = bnt[c];
  #pragma unroll
  for (int k = 0; k < NTY; ++k) s += atype[row * NTY + k] * Wnt[k * NH + c];
  typeG[idx] = s;
}

__global__ void enc_agent_kernel(const float* __restrict__ adata, const float* __restrict__ rel,
                                 const float* __restrict__ Wa, const float* __restrict__ ba,
                                 float* __restrict__ agentG) {
  int idx = blockIdx.x * 256 + threadIdx.x;              // (b,a,t,c)
  if (idx >= NB * NA * NT * NH) return;
  int c = idx & 127, row = idx >> 7;
  float s = ba[c];
  #pragma unroll
  for (int k = 0; k < NAG; ++k) s += adata[row * NAG + k] * Wa[k * NH + c];
  agentG[idx] = s * rel[row / NT];
}

__global__ void enc_scene_kernel(const float* __restrict__ scene,
                                 const float* __restrict__ Ws, const float* __restrict__ bs,
                                 float* __restrict__ sceneG) {
  int idx = blockIdx.x * 256 + threadIdx.x;              // (b,t,c)
  if (idx >= NB * NT * NH) return;
  int c = idx & 127, row = idx >> 7;
  float s = bs[c];
  #pragma unroll
  for (int k = 0; k < NSC; ++k) s += scene[row * NSC + k] * Ws[k * NH + c];
  sceneG[idx] = s;
}

__global__ void enc_edge0_kernel(const float* __restrict__ ntraj, const float* __restrict__ traj,
                                 const float* __restrict__ atype,
                                 const float* __restrict__ Wein, const float* __restrict__ bein,
                                 const float* __restrict__ Wety, const float* __restrict__ bety,
                                 h16* __restrict__ esH) {
  int idx = blockIdx.x * 256 + threadIdx.x;              // (b,i,j,c)
  if (idx >= NB * NA * NA * NE) return;
  int c = idx & 127;
  int e = idx >> 7;
  int j = e & 63, i = (e >> 6) & 63, b = e >> 12;
  const float* ni  = ntraj + (size_t)((b * NA + i) * NT) * 3;  // t = 0
  const float* ti  = traj  + (size_t)((b * NA + i) * NT) * 3;
  const float* nj  = ntraj + (size_t)((b * NA + j) * NT) * 3;
  const float* tj  = traj  + (size_t)((b * NA + j) * NT) * 3;
  const float* tyi = atype + (size_t)(b * NA + i) * NTY;
  const float* tyj = atype + (size_t)(b * NA + j) * NTY;
  float s = bein[c] + bety[c];
  #pragma unroll
  for (int k = 0; k < 3; ++k) s += ni[k] * Wein[k * NE + c];
  #pragma unroll
  for (int k = 0; k < 3; ++k) s += ti[k] * Wein[(3 + k) * NE + c];
  #pragma unroll
  for (int k = 0; k < 3; ++k) s += nj[k] * Wein[(6 + k) * NE + c];
  #pragma unroll
  for (int k = 0; k < 3; ++k) s += tj[k] * Wein[(9 + k) * NE + c];
  #pragma unroll
  for (int k = 0; k < NTY; ++k) s += tyi[k] * (Wein[(12 + k) * NE + c] + Wety[k * NE + c]);
  #pragma unroll
  for (int k = 0; k < NTY; ++k) s += tyj[k] * (Wein[(20 + k) * NE + c] + Wety[(NTY + k) * NE + c]);
  esH[idx] = (h16)s;
}

// ---------------------------------------------------------------------------
// Edge kernel: one workgroup per (b,i).  es_new[b,i,j,:] = [h_i|h_j|es] @ W_edge + b
// es rows (already f16 in global) are staged into LDS with the CDNA5 async
// global->LDS DMA path (ASYNCcnt), overlapping the f32->f16 conversion of
// h_i/h_j done on the VALU.  agg[b,i,:] = sum_j es_new reduced in-register.
// ---------------------------------------------------------------------------

__global__ __launch_bounds__(256) void edge_kernel(
    const float* __restrict__ hG,        // [B][A][H]
    h16* __restrict__ esH,               // [B][A][A][E] f16 state (in/out)
    float* __restrict__ aggE,            // [B][A][E]
    const h16* __restrict__ WeT,         // [128][384] (N-major)
    const float* __restrict__ b_edge) {
  extern __shared__ char smem[];
  h16* Alds = (h16*)smem;                // 64 x SEDGE halfs
  int i = blockIdx.x, b = blockIdx.y;
  int tid = threadIdx.x;
  const float* hB = hG + (size_t)b * NA * NH;
  const float* hi = hB + (size_t)i * NH;

  __builtin_prefetch((const void*)WeT, 0, 1);

  int j = tid >> 2;                      // row 0..63, 4 threads per row
  int q = tid & 3;

  // Async DMA: es[b,i,j,:] (256 B/row) -> LDS cols [256:384), no VGPR bounce.
  // global_load_async_to_lds_b128: VDST = LDS byte address, VADDR = 64-bit
  // global address.  Tracked by ASYNCcnt.
  {
    const h16* esr = esH + ((((size_t)b * NA + i) * NA + j) * NE);
    unsigned ldsb = (unsigned)(size_t)(Alds + j * SEDGE + 2 * NH);
    #pragma unroll
    for (int ck = 0; ck < 4; ++ck) {
      unsigned       ldst = ldsb + (unsigned)((q * 4 + ck) * 16);
      unsigned long long ga = (unsigned long long)(size_t)(esr + (q * 4 + ck) * 8);
      asm volatile("global_load_async_to_lds_b128 %0, %1, off"
                   :: "v"(ldst), "v"(ga) : "memory");
    }
  }

  // Meanwhile: convert h_i (broadcast) and h_j rows to f16 in LDS cols [0:256).
  {
    int part = q * 32;
    const float* hj = hB + (size_t)j * NH;
    h16* dst = Alds + j * SEDGE;
    for (int c = part; c < part + 32; ++c) {
      dst[c]      = (h16)hi[c];
      dst[NH + c] = (h16)hj[c];
    }
  }
  asm volatile("s_wait_asynccnt 0x0" ::: "memory");  // async LDS writes landed
  __syncthreads();

  int w = tid >> 5;                      // wave -> N-tile (cols 16w..16w+15)
  v8f zero = {};
  v8f acc[4];
  #pragma unroll
  for (int mt = 0; mt < 4; ++mt) acc[mt] = zero;

  for (int kt = 0; kt < KEDGE / 32; ++kt) {
    v16h bf = load_b_glb(WeT, KEDGE, w * 16, kt * 32);
    #pragma unroll
    for (int mt = 0; mt < 4; ++mt) {
      v16h af = load_a_lds(Alds, SEDGE, mt * 16, kt * 32);
      acc[mt] = WMMA(af, bf, acc[mt]);
    }
  }

  // Epilogue: +bias, store es (f16 state), reduce over rows for agg.
  int lane = tid & 31;
  int col  = w * 16 + (lane & 15);
  float bias = b_edge[col];
  float colsum = 0.f;
  #pragma unroll
  for (int mt = 0; mt < 4; ++mt) {
    #pragma unroll
    for (int r = 0; r < 8; ++r) {
      float v = acc[mt][r] + bias;
      int row = mt * 16 + r + ((lane >> 4) << 3);
      esH[(((size_t)b * NA + i) * NA + row) * NE + col] = (h16)v;
      colsum += v;
    }
  }
  colsum += __shfl_xor(colsum, 16, 32);
  if (lane < 16) aggE[((size_t)b * NA + i) * NH + col] = colsum;
}

// ---------------------------------------------------------------------------
// Node kernel: one workgroup per batch b.  se_new (scalar) -> agg ->
// e2n WMMA -> gates WMMA (768x512) -> LSTM -> pred WMMA -> ReLU out.
// ---------------------------------------------------------------------------

__global__ __launch_bounds__(256) void node_kernel(
    float* __restrict__ hG, float* __restrict__ cG, float* __restrict__ seG,
    const float* __restrict__ aggE,
    const float* __restrict__ coordG, const float* __restrict__ typeG,
    const float* __restrict__ sceneG, const float* __restrict__ agentG,
    const float* __restrict__ atype,
    const float* __restrict__ W_self, const float* __restrict__ b_self,
    const h16* __restrict__ We2nT, const float* __restrict__ b_e2n,
    const h16* __restrict__ WgT,
    const float* __restrict__ b_ih, const float* __restrict__ b_hh,
    const h16* __restrict__ WpT, const float* __restrict__ b_pred,
    float* __restrict__ out, int t) {
  extern __shared__ char smem[];
  float* hF   = (float*)(smem);                          // 64x128 f32   (32768 B)
  float* agg  = (float*)(smem + 32768);                  // 64x128 f32   (32768 B)
  h16*   xrow = (h16*)(smem + 65536);                    // 64xSX  f16   (99328 B)
  h16*   aggH = (h16*)(smem + 65536 + 99328);            // 64xSH  f16   (17408 B)
  h16*   hH   = (h16*)(smem + 65536 + 99328 + 17408);    // 64xSH  f16   (17408 B)

  int b = blockIdx.x, tid = threadIdx.x;
  int w = tid >> 5, lane = tid & 31;
  const size_t bOff = (size_t)b * NA * NH;

  // Phase 0: stage old h; scalar se_new into agg (as temp).
  for (int idx = tid; idx < NA * NH; idx += 256) hF[idx] = hG[bOff + idx];
  __syncthreads();
  {
    int a  = tid >> 2;
    int c0 = (tid & 3) * 32;
    const float* ty = atype + (size_t)(b * NA + a) * NTY;
    const float* se = seG + bOff + (size_t)a * NH;
    for (int c = c0; c < c0 + 32; ++c) {
      float s = b_self[c];
      for (int k = 0; k < NH; ++k)  s += hF[a * NH + k] * W_self[k * NE + c];
      for (int k = 0; k < NE; ++k)  s += se[k] * W_self[(NH + k) * NE + c];
      #pragma unroll
      for (int k = 0; k < NTY; ++k) s += ty[k] * W_self[(NH + NE + k) * NE + c];
      agg[a * NH + c] = s;                               // temp = se_new
    }
  }
  __syncthreads();

  // Phase 1: commit se state, form agg(+edges) in f16; fill static xrow cols.
  for (int idx = tid; idx < NA * NH; idx += 256) {
    int a = idx >> 7, c = idx & 127;
    float sv = agg[idx];
    seG[bOff + idx] = sv;
    float av = sv + aggE[bOff + idx];
    aggH[a * SH + c] = (h16)av;
    h16* x = xrow + a * SX;
    x[c]            = (h16)coordG[((size_t)(b * NA + a) * NT + t) * NH + c];
    x[NH + c]       = (h16)typeG[(size_t)(b * NA + a) * NH + c];
    x[3 * NH + c]   = (h16)sceneG[((size_t)b * NT + t) * NH + c];
    x[4 * NH + c]   = (h16)agentG[((size_t)(b * NA + a) * NT + t) * NH + c];
    x[5 * NH + c]   = (h16)hF[idx];
  }
  __syncthreads();

  v8f zero = {};
  float tscale = (t > 0) ? 1.f : 0.f;

  // Phase 2: node_edges = (agg @ W_e2n + b) * (t>0)  -> xrow[:, 256:384]
  {
    v8f acc[4];
    #pragma unroll
    for (int mt = 0; mt < 4; ++mt) acc[mt] = zero;
    for (int kt = 0; kt < 4; ++kt) {
      v16h bf = load_b_glb(We2nT, NH, w * 16, kt * 32);
      #pragma unroll
      for (int mt = 0; mt < 4; ++mt) {
        v16h af = load_a_lds(aggH, SH, mt * 16, kt * 32);
        acc[mt] = WMMA(af, bf, acc[mt]);
      }
    }
    int col = w * 16 + (lane & 15);
    float bias = b_e2n[col];
    #pragma unroll
    for (int mt = 0; mt < 4; ++mt) {
      #pragma unroll
      for (int r = 0; r < 8; ++r) {
        int row = mt * 16 + r + ((lane >> 4) << 3);
        xrow[row * SX + 2 * NH + col] = (h16)((acc[mt][r] + bias) * tscale);
      }
    }
  }
  __syncthreads();

  // Phase 3: gates = [x|h] @ [W_ih;W_hh]^T-combined (K=768, N=512).
  // Wave w owns gate N-tiles {w, w+8, w+16, w+24}: i/f/g/o for the SAME
  // 16 hidden units -> LSTM nonlinearity entirely in-register.
  {
    v8f gacc[4][4];
    #pragma unroll
    for (int g = 0; g < 4; ++g)
      #pragma unroll
      for (int mt = 0; mt < 4; ++mt) gacc[g][mt] = zero;

    for (int kt = 0; kt < KG / 32; ++kt) {
      v16h af[4];
      #pragma unroll
      for (int mt = 0; mt < 4; ++mt) af[mt] = load_a_lds(xrow, SX, mt * 16, kt * 32);
      #pragma unroll
      for (int g = 0; g < 4; ++g) {
        v16h bf = load_b_glb(WgT, KG, g * 128 + w * 16, kt * 32);
        #pragma unroll
        for (int mt = 0; mt < 4; ++mt) gacc[g][mt] = WMMA(af[mt], bf, gacc[g][mt]);
      }
    }

    int nloc = w * 16 + (lane & 15);                     // hidden unit 0..127
    float bi  = b_ih[nloc]       + b_hh[nloc];
    float bf_ = b_ih[128 + nloc] + b_hh[128 + nloc];
    float bg  = b_ih[256 + nloc] + b_hh[256 + nloc];
    float bo  = b_ih[384 + nloc] + b_hh[384 + nloc];
    #pragma unroll
    for (int mt = 0; mt < 4; ++mt) {
      #pragma unroll
      for (int r = 0; r < 8; ++r) {
        int row = mt * 16 + r + ((lane >> 4) << 3);      // agent a
        float gi = gacc[0][mt][r] + bi;
        float gf = gacc[1][mt][r] + bf_;
        float gg = gacc[2][mt][r] + bg;
        float go = gacc[3][mt][r] + bo;
        size_t sidx = bOff + (size_t)row * NH + nloc;
        float cn = sigmf(gf) * cG[sidx] + sigmf(gi) * tanhf(gg);
        float hn = sigmf(go) * tanhf(cn);
        cG[sidx] = cn;
        hG[sidx] = hn;
        hH[row * SH + nloc] = (h16)hn;
      }
    }
  }
  __syncthreads();

  // Phase 4: out = relu(h_new @ W_pred + b_pred)
  {
    v8f acc[4];
    #pragma unroll
    for (int mt = 0; mt < 4; ++mt) acc[mt] = zero;
    for (int kt = 0; kt < 4; ++kt) {
      v16h bf = load_b_glb(WpT, NH, w * 16, kt * 32);
      #pragma unroll
      for (int mt = 0; mt < 4; ++mt) {
        v16h af = load_a_lds(hH, SH, mt * 16, kt * 32);
        acc[mt] = WMMA(af, bf, acc[mt]);
      }
    }
    int col = w * 16 + (lane & 15);
    float bias = b_pred[col];
    #pragma unroll
    for (int mt = 0; mt < 4; ++mt) {
      #pragma unroll
      for (int r = 0; r < 8; ++r) {
        int row = mt * 16 + r + ((lane >> 4) << 3);
        float v = fmaxf(acc[mt][r] + bias, 0.f);
        out[((size_t)(b * NA + row) * NT + t) * NH + col] = v;   // (B,A,T,O)
      }
    }
  }
}

// ---------------------------------------------------------------------------
// Host launcher
// ---------------------------------------------------------------------------

extern "C" void kernel_launch(void* const* d_in, const int* in_sizes, int n_in,
                              void* d_out, int out_size, void* d_ws, size_t ws_size,
                              hipStream_t stream) {
  (void)in_sizes; (void)n_in; (void)out_size; (void)ws_size;
  const float* traj   = (const float*)d_in[0];
  const float* ntraj  = (const float*)d_in[1];
  const float* atype  = (const float*)d_in[2];
  const float* scene  = (const float*)d_in[3];
  const float* adata  = (const float*)d_in[4];
  const float* relev  = (const float*)d_in[5];
  const float* W_in   = (const float*)d_in[6];
  const float* b_in   = (const float*)d_in[7];
  const float* W_nt   = (const float*)d_in[8];
  const float* b_nt   = (const float*)d_in[9];
  const float* W_ag   = (const float*)d_in[10];
  const float* b_ag   = (const float*)d_in[11];
  const float* W_sc   = (const float*)d_in[12];
  const float* b_sc   = (const float*)d_in[13];
  const float* W_ein  = (const float*)d_in[14];
  const float* b_ein  = (const float*)d_in[15];
  const float* W_ety  = (const float*)d_in[16];
  const float* b_ety  = (const float*)d_in[17];
  const float* W_edge = (const float*)d_in[18];
  const float* b_edge = (const float*)d_in[19];
  const float* W_self = (const float*)d_in[20];
  const float* b_self = (const float*)d_in[21];
  const float* W_e2n  = (const float*)d_in[22];
  const float* b_e2n  = (const float*)d_in[23];
  const float* W_ih   = (const float*)d_in[24];
  const float* W_hh   = (const float*)d_in[25];
  const float* b_ih   = (const float*)d_in[26];
  const float* b_hh   = (const float*)d_in[27];
  const float* W_pred = (const float*)d_in[28];
  const float* b_pred = (const float*)d_in[29];
  float* out = (float*)d_out;

  // Workspace carve (bump allocator, 256B aligned)
  char* ws = (char*)d_ws;
  auto alloc = [&](size_t bytes) {
    char* p = ws;
    ws += (bytes + 255) & ~(size_t)255;
    return p;
  };
  h16*   esH    = (h16*)  alloc((size_t)NB * NA * NA * NE * sizeof(h16));   // 8 MB
  float* hG     = (float*)alloc((size_t)NB * NA * NH * sizeof(float));
  float* cG     = (float*)alloc((size_t)NB * NA * NH * sizeof(float));
  float* seG    = (float*)alloc((size_t)NB * NA * NH * sizeof(float));
  float* aggE   = (float*)alloc((size_t)NB * NA * NH * sizeof(float));
  float* coordG = (float*)alloc((size_t)NB * NA * NT * NH * sizeof(float));
  float* typeG  = (float*)alloc((size_t)NB * NA * NH * sizeof(float));
  float* agentG = (float*)alloc((size_t)NB * NA * NT * NH * sizeof(float));
  float* sceneG = (float*)alloc((size_t)NB * NT * NH * sizeof(float));
  h16*   WeT    = (h16*)  alloc((size_t)NE * KEDGE * sizeof(h16));
  h16*   WgT    = (h16*)  alloc((size_t)512 * KG * sizeof(h16));
  h16*   We2nT  = (h16*)  alloc((size_t)NH * NH * sizeof(h16));
  h16*   WpT    = (h16*)  alloc((size_t)NH * NH * sizeof(h16));

  auto blocks = [](int n) { return (n + 255) / 256; };

  // Setup (re-run every call: deterministic, no cross-call state)
  zero_state_kernel<<<blocks(NB * NA * NH), 256, 0, stream>>>(hG, cG, seG);
  transpose_to_h16<<<blocks(KEDGE * NE), 256, 0, stream>>>(W_edge, WeT, KEDGE, NE);
  transpose_to_h16<<<blocks(NH * NH), 256, 0, stream>>>(W_e2n, We2nT, NH, NH);
  transpose_to_h16<<<blocks(NH * NH), 256, 0, stream>>>(W_pred, WpT, NH, NH);
  build_wgT<<<blocks(512 * KG), 256, 0, stream>>>(W_ih, W_hh, WgT);
  enc_coord_kernel<<<blocks(NB * NA * NT * NH), 256, 0, stream>>>(ntraj, traj, W_in, b_in, coordG);
  enc_type_kernel<<<blocks(NB * NA * NH), 256, 0, stream>>>(atype, W_nt, b_nt, typeG);
  enc_agent_kernel<<<blocks(NB * NA * NT * NH), 256, 0, stream>>>(adata, relev, W_ag, b_ag, agentG);
  enc_scene_kernel<<<blocks(NB * NT * NH), 256, 0, stream>>>(scene, W_sc, b_sc, sceneG);
  enc_edge0_kernel<<<blocks(NB * NA * NA * NE), 256, 0, stream>>>(
      ntraj, traj, atype, W_ein, b_ein, W_ety, b_ety, esH);

  const size_t edgeLds = (size_t)NA * SEDGE * sizeof(h16);            // ~50 KB
  const size_t nodeLds = 32768 + 32768 + (size_t)NA * SX * 2 + (size_t)NA * SH * 2 * 2; // ~195 KB

  // Sequential recurrence: edge update + node/LSTM update per timestep.
  for (int t = 0; t < NT; ++t) {
    edge_kernel<<<dim3(NA, NB), 256, edgeLds, stream>>>(hG, esH, aggE, WeT, b_edge);
    node_kernel<<<dim3(NB), 256, nodeLds, stream>>>(
        hG, cG, seG, aggE, coordG, typeG, sceneG, agentG, atype,
        W_self, b_self, We2nT, b_e2n, WgT, b_ih, b_hh, WpT, b_pred, out, t);
  }
}
